// KalmanNetQ_283467842224
// MI455X (gfx1250) — matrix operand order repaired and checked by reading, not compiled, not run on hardware
//
#include <hip/hip_runtime.h>
#include <math.h>

#define S_LEN 2048
#define HID   64
#define NNIN  13
#define GATES 192
#define BT    16     // batch tile per wave (WMMA M)
#define DT_   0.01f
#define LOG2E 1.4426950408889634f

typedef __attribute__((ext_vector_type(16))) __bf16 bf16x16;
typedef __attribute__((ext_vector_type(8)))  float  f32x8;

union Frag16 { bf16x16 v; uint4 q[2]; };

static __device__ __forceinline__ unsigned short f2bf(float f) {
    unsigned u = __float_as_uint(f);
    u += 0x7FFFu + ((u >> 16) & 1u);          // round-to-nearest-even
    return (unsigned short)(u >> 16);
}

static __device__ __forceinline__ f32x8 wmma_bf16(const Frag16 a, const Frag16 b, f32x8 c) {
    return __builtin_amdgcn_wmma_f32_16x16x32_bf16(false, a.v, false, b.v,
                                                   (short)0, c, false, false);
}

// Fast HW transcendentals: v_exp_f32 / v_rcp_f32 / v_rsq_f32, no libm expansion.
static __device__ __forceinline__ float fexp2_(float x) { return __builtin_amdgcn_exp2f(x); }
static __device__ __forceinline__ float frcp_(float x)  { return __builtin_amdgcn_rcpf(x); }
static __device__ __forceinline__ float frsq_(float x)  { return __builtin_amdgcn_rsqf(x); }
static __device__ __forceinline__ float sigm_(float x) {
    return frcp_(1.0f + fexp2_(-LOG2E * x));
}
static __device__ __forceinline__ float tanh_(float x) {
    float e = fexp2_(-2.0f * LOG2E * x);      // e^(-2x)
    return (1.0f - e) * frcp_(1.0f + e);
}

// A fragment (16x32 bf16, row-major source): lane<16 -> M=lane, K {kb..kb+7, kb+16..kb+23}
//                                            lane>=16 -> M=lane-16, K shifted by +8
static __device__ __forceinline__ Frag16 load_a(const unsigned short* base, int rowStride,
                                                int kbase, int lane) {
    int m  = lane & 15;
    int kb = kbase + ((lane & 16) ? 8 : 0);
    const unsigned short* p = base + m * rowStride + kb;
    Frag16 f;
    f.q[0] = *(const uint4*)(p);        // K kb..kb+7
    f.q[1] = *(const uint4*)(p + 16);   // K kb+16..kb+23
    return f;
}

// B fragment (32x16 bf16): lane n holds column col+(lane&15), K {kb..kb+7, kb+16..kb+23}
// (kb = kbase for lanes 0-15, kbase+8 for lanes 16-31).
// Source W is row-major [feature][k], so B[k][n] = W[n][k] -> read row `col`.
static __device__ __forceinline__ Frag16 load_b(const unsigned short* base, int rowStride,
                                                int col, int kbase, int lane) {
    int n  = col + (lane & 15);
    int kb = kbase + ((lane & 16) ? 8 : 0);
    const unsigned short* p = base + n * rowStride + kb;
    Frag16 f;
    f.q[0] = *(const uint4*)(p);
    f.q[1] = *(const uint4*)(p + 16);
    return f;
}

__global__ __launch_bounds__(32)
void kalmannet_gru_scan(const float* __restrict__ inputs,
                        const float* __restrict__ ln_w, const float* __restrict__ ln_b,
                        const float* __restrict__ w_ih, const float* __restrict__ w_hh,
                        const float* __restrict__ b_ih, const float* __restrict__ b_hh,
                        const float* __restrict__ fc_w, const float* __restrict__ fc_b,
                        float* __restrict__ out)
{
    // ---- LDS: weight staging (used once) + per-step staging buffers ------
    __shared__ __align__(16) unsigned short Wi[GATES][32];   // w_ih, K padded 13->32
    __shared__ __align__(16) unsigned short Wh[GATES][HID];  // w_hh
    __shared__ __align__(16) unsigned short Wf[32][HID];     // fc_w, rows padded 24->32
    __shared__ __align__(16) unsigned short Xl[BT][32];      // gru_in staging (bf16)
    __shared__ __align__(16) unsigned short Hl[BT][HID];     // h staging (bf16)
    __shared__ __align__(16) float Fo[BT][32];               // fc output staging
    __shared__ float BRZ[128];                               // b_ih+b_hh for r,z gates
    __shared__ float BIN[64], BHN[64];                       // n-gate biases (separate)
    __shared__ float FB[24], LNW[16], LNB[16];

    const int lane = threadIdx.x;
    const int wg   = blockIdx.x;
    const int colf = lane & 15;
    const int hi8  = (lane & 16) ? 8 : 0;

    // ---- one-time weight load / LDS init --------------------------------
    for (int i = lane; i < GATES * 32; i += 32) {
        int n = i >> 5, k = i & 31;
        Wi[n][k] = (k < NNIN) ? f2bf(w_ih[n * NNIN + k]) : (unsigned short)0;
    }
    for (int i = lane; i < GATES * HID; i += 32) {
        int n = i >> 6, k = i & 63;
        Wh[n][k] = f2bf(w_hh[n * HID + k]);
    }
    for (int i = lane; i < 32 * HID; i += 32) {
        int n = i >> 6, k = i & 63;
        Wf[n][k] = (n < 24) ? f2bf(fc_w[n * HID + k]) : (unsigned short)0;
    }
    for (int i = lane; i < 128; i += 32) BRZ[i] = b_ih[i] + b_hh[i];
    for (int i = lane; i < 64;  i += 32) { BIN[i] = b_ih[128 + i]; BHN[i] = b_hh[128 + i]; }
    if (lane < 24) FB[lane] = fc_b[lane];
    if (lane < 16) { LNW[lane] = (lane < NNIN) ? ln_w[lane] : 0.f;
                     LNB[lane] = (lane < NNIN) ? ln_b[lane] : 0.f; }
    for (int i = lane; i < BT * 32;  i += 32) ((unsigned short*)Xl)[i] = 0;
    for (int i = lane; i < BT * HID; i += 32) ((unsigned short*)Hl)[i] = 0;
    __syncthreads();

    // ---- hoist ALL loop-invariant weight fragments into registers --------
    // 40 fragments x 8 VGPRs = 320 VGPRs; wave32 budget is 1024 and we run
    // one wave per SIMD, so register pressure beats per-step LDS latency.
    Frag16 Bi[3][4];        // [gate r/z/n][n-tile], w_ih (single K-step)
    Frag16 Bh[3][4][2];     // [gate][n-tile][K-step], w_hh
    Frag16 Bf[2][2];        // [n-tile][K-step], fc_w
    #pragma unroll
    for (int g = 0; g < 3; ++g) {
        #pragma unroll
        for (int t4 = 0; t4 < 4; ++t4) {
            Bi[g][t4]    = load_b(&Wi[0][0], 32,  g * 64 + t4 * 16, 0,  lane);
            Bh[g][t4][0] = load_b(&Wh[0][0], HID, g * 64 + t4 * 16, 0,  lane);
            Bh[g][t4][1] = load_b(&Wh[0][0], HID, g * 64 + t4 * 16, 32, lane);
        }
    }
    #pragma unroll
    for (int ft = 0; ft < 2; ++ft) {
        Bf[ft][0] = load_b(&Wf[0][0], HID, ft * 16, 0,  lane);
        Bf[ft][1] = load_b(&Wf[0][0], HID, ft * 16, 32, lane);
    }
    // per-lane gate biases (registers)
    float brzR[4], brzZ[4], binN[4], bhnN[4];
    #pragma unroll
    for (int t4 = 0; t4 < 4; ++t4) {
        brzR[t4] = BRZ[t4 * 16 + colf];
        brzZ[t4] = BRZ[64 + t4 * 16 + colf];
        binN[t4] = BIN[t4 * 16 + colf];
        bhnN[t4] = BHN[t4 * 16 + colf];
    }
    float fbv[2];
    #pragma unroll
    for (int ft = 0; ft < 2; ++ft) {
        int c = ft * 16 + colf;
        fbv[ft] = (c < 24) ? FB[c] : 0.f;    // fc bias folded into Fo store
    }

    // ---- persistent per-lane EKF state (lanes 0..15, lane == batch-in-tile)
    float p0 = 0.f, p1 = 0.f, p2 = 0.f;
    float v0 = 0.f, v1 = 0.f, v2 = 0.f;
    float qw = 1.f, qx = 0.f, qy = 0.f, qz = 0.f;

    // h in D layout: hD[tile][i] = h[batchrow, tile*16 + colf]
    f32x8 hD[4];
    #pragma unroll
    for (int t4 = 0; t4 < 4; ++t4) hD[t4] = (f32x8){0.f,0.f,0.f,0.f,0.f,0.f,0.f,0.f};

    const int    bg   = wg * BT + colf;                 // global batch index (lanes 0..15)
    const float* yrow = inputs + (size_t)bg * S_LEN * 9;
    float*       orow = out    + (size_t)bg * S_LEN * 10;

    // A fragments of previous h (zero at t=0, refreshed at end of each step)
    Frag16 Ah0 = load_a(&Hl[0][0], HID, 0,  lane);
    Frag16 Ah1 = load_a(&Hl[0][0], HID, 32, lane);

    #pragma unroll 1
    for (int t = 0; t < S_LEN; ++t) {
        float qpw, qpx, qpy, qpz;                 // q_pred
        float pp0, pp1, pp2, vp0, vp1, vp2;       // p_pred, v_pred
        float inn[6];

        // ================= E1: physics + LayerNorm (lane == batch) =========
        if (lane < BT) {
            const float* yp = yrow + (size_t)t * 9;
            __builtin_prefetch(yp + 9, 0, 1);     // global_prefetch_b8 for t+1
            float ax = yp[0], ay = yp[1], az = yp[2];
            float gx = yp[3], gy = yp[4], gz = yp[5];
            float mx = yp[6], my = yp[7], mz = yp[8];

            // dq = norm([1, gyro*DT/2]); q_pred = norm(q * dq)
            float hx = gx * (0.5f * DT_), hy = gy * (0.5f * DT_), hz = gz * (0.5f * DT_);
            float dn = frsq_(1.f + hx * hx + hy * hy + hz * hz);
            float dw = dn; hx *= dn; hy *= dn; hz *= dn;
            qpw = qw * dw - qx * hx - qy * hy - qz * hz;
            qpx = qw * hx + qx * dw + qy * hz - qz * hy;
            qpy = qw * hy - qx * hz + qy * dw + qz * hx;
            qpz = qw * hz + qx * hy - qy * hx + qz * dw;
            float qn = frsq_(qpw * qpw + qpx * qpx + qpy * qpy + qpz * qpz);
            qpw *= qn; qpx *= qn; qpy *= qn; qpz *= qn;

            // accel_world = R(q) @ accel - g
            float xx = qx * qx, yy = qy * qy, zz = qz * qz;
            float xy = qx * qy, xz = qx * qz, yz = qy * qz;
            float wx = qw * qx, wy = qw * qy, wz = qw * qz;
            float aw0 = (1.f - 2.f * (yy + zz)) * ax + 2.f * (xy - wz) * ay + 2.f * (xz + wy) * az;
            float aw1 = 2.f * (xy + wz) * ax + (1.f - 2.f * (xx + zz)) * ay + 2.f * (yz - wx) * az;
            float aw2 = 2.f * (xz - wy) * ax + 2.f * (yz + wx) * ay + (1.f - 2.f * (xx + yy)) * az
                        - 9.81f;
            vp0 = v0 + aw0 * DT_;  vp1 = v1 + aw1 * DT_;  vp2 = v2 + aw2 * DT_;
            pp0 = p0 + v0 * DT_ + 0.5f * aw0 * DT_ * DT_;
            pp1 = p1 + v1 * DT_ + 0.5f * aw1 * DT_ * DT_;
            pp2 = p2 + v2 * DT_ + 0.5f * aw2 * DT_ * DT_;

            // Rp = R(q_pred); body-frame references: g_body_i = Rp[2][i]*9.81,
            // mag_body_i = 20*Rp[0][i] - 40*Rp[2][i]
            float pxx = qpx * qpx, pyy = qpy * qpy, pzz = qpz * qpz;
            float pxy = qpx * qpy, pxz = qpx * qpz, pyz = qpy * qpz;
            float pwx = qpw * qpx, pwy = qpw * qpy, pwz = qpw * qpz;
            float r00 = 1.f - 2.f * (pyy + pzz), r01 = 2.f * (pxy - pwz), r02 = 2.f * (pxz + pwy);
            float r20 = 2.f * (pxz - pwy), r21 = 2.f * (pyz + pwx), r22 = 1.f - 2.f * (pxx + pyy);
            inn[0] = mx - (20.f * r00 - 40.f * r20);
            inn[1] = my - (20.f * r01 - 40.f * r21);
            inn[2] = mz - (20.f * r02 - 40.f * r22);
            inn[3] = ax - 9.81f * r20;
            inn[4] = ay - 9.81f * r21;
            inn[5] = az - 9.81f * r22;

            // LayerNorm over 13-vector [innov, v_pred, q_pred]
            float gin[NNIN] = { inn[0], inn[1], inn[2], inn[3], inn[4], inn[5],
                                vp0, vp1, vp2, qpw, qpx, qpy, qpz };
            float s = 0.f;
            #pragma unroll
            for (int k = 0; k < NNIN; ++k) s += gin[k];
            float mean = s * (1.0f / NNIN);
            float var = 0.f;
            #pragma unroll
            for (int k = 0; k < NNIN; ++k) { float d = gin[k] - mean; var += d * d; }
            float inv = frsq_(var * (1.0f / NNIN) + 1e-5f);
            #pragma unroll
            for (int k = 0; k < NNIN; ++k)
                Xl[lane][k] = f2bf((gin[k] - mean) * inv * LNW[k] + LNB[k]);
        }
        __syncthreads();

        // ================= M1+E2: GRU cell on WMMA (all 32 lanes) ==========
        Frag16 Ax = load_a(&Xl[0][0], 32, 0, lane);
        #pragma unroll
        for (int t4 = 0; t4 < 4; ++t4) {
            f32x8 aR = (f32x8){0.f,0.f,0.f,0.f,0.f,0.f,0.f,0.f};
            aR = wmma_bf16(Ax,  Bi[0][t4],    aR);
            aR = wmma_bf16(Ah0, Bh[0][t4][0], aR);
            aR = wmma_bf16(Ah1, Bh[0][t4][1], aR);
            f32x8 aZ = (f32x8){0.f,0.f,0.f,0.f,0.f,0.f,0.f,0.f};
            aZ = wmma_bf16(Ax,  Bi[1][t4],    aZ);
            aZ = wmma_bf16(Ah0, Bh[1][t4][0], aZ);
            aZ = wmma_bf16(Ah1, Bh[1][t4][1], aZ);
            // n gate: input and hidden parts must stay separate (r scales only h_n)
            f32x8 aGN = (f32x8){0.f,0.f,0.f,0.f,0.f,0.f,0.f,0.f};
            aGN = wmma_bf16(Ax,  Bi[2][t4],    aGN);
            f32x8 aHN = (f32x8){0.f,0.f,0.f,0.f,0.f,0.f,0.f,0.f};
            aHN = wmma_bf16(Ah0, Bh[2][t4][0], aHN);
            aHN = wmma_bf16(Ah1, Bh[2][t4][1], aHN);

            #pragma unroll
            for (int i = 0; i < 8; ++i) {
                float r  = sigm_(aR[i] + brzR[t4]);
                float z  = sigm_(aZ[i] + brzZ[t4]);
                float n  = tanh_(aGN[i] + binN[t4] + r * (aHN[i] + bhnN[t4]));
                float hn = (1.f - z) * n + z * hD[t4][i];
                hD[t4][i] = hn;
                Hl[i + hi8][t4 * 16 + colf] = f2bf(hn);   // new h -> LDS (bf16)
            }
        }
        __syncthreads();

        // ================= M2: fc (Kalman gain) on new h ===================
        Ah0 = load_a(&Hl[0][0], HID, 0,  lane);   // reused for fc AND next-step gh
        Ah1 = load_a(&Hl[0][0], HID, 32, lane);
        #pragma unroll
        for (int ft = 0; ft < 2; ++ft) {
            f32x8 aF = (f32x8){0.f,0.f,0.f,0.f,0.f,0.f,0.f,0.f};
            aF = wmma_bf16(Ah0, Bf[ft][0], aF);
            aF = wmma_bf16(Ah1, Bf[ft][1], aF);
            #pragma unroll
            for (int i = 0; i < 8; ++i)
                Fo[i + hi8][ft * 16 + colf] = aF[i] + fbv[ft];   // fc bias folded in
        }
        __syncthreads();

        // ================= E3: Kalman correction + emit (lane == batch) ====
        if (lane < BT) {
            float dq[4];
            #pragma unroll
            for (int i = 0; i < 4; ++i) {
                float s = 0.f;
                #pragma unroll
                for (int j = 0; j < 6; ++j)
                    s += Fo[lane][i * 6 + j] * inn[j];
                dq[i] = s;
            }
            float f0 = qpw + dq[0], f1 = qpx + dq[1], f2 = qpy + dq[2], f3 = qpz + dq[3];
            float qn = frsq_(f0 * f0 + f1 * f1 + f2 * f2 + f3 * f3);
            qw = f0 * qn; qx = f1 * qn; qy = f2 * qn; qz = f3 * qn;
            p0 = pp0; p1 = pp1; p2 = pp2;
            v0 = vp0; v1 = vp1; v2 = vp2;

            float* op = orow + (size_t)t * 10;
            op[0] = pp0; op[1] = pp1; op[2] = pp2;
            op[3] = vp0; op[4] = vp1; op[5] = vp2;
            op[6] = qw;  op[7] = qx;  op[8] = qy;  op[9] = qz;
        }
        __syncthreads();
    }
}

extern "C" void kernel_launch(void* const* d_in, const int* in_sizes, int n_in,
                              void* d_out, int out_size, void* d_ws, size_t ws_size,
                              hipStream_t stream) {
    const float* inputs = (const float*)d_in[0];
    const float* ln_w   = (const float*)d_in[1];
    const float* ln_b   = (const float*)d_in[2];
    const float* w_ih   = (const float*)d_in[3];
    const float* w_hh   = (const float*)d_in[4];
    const float* b_ih   = (const float*)d_in[5];
    const float* b_hh   = (const float*)d_in[6];
    const float* fc_w   = (const float*)d_in[7];
    const float* fc_b   = (const float*)d_in[8];
    float* out = (float*)d_out;

    const int B = in_sizes[0] / (S_LEN * 9);     // 512
    dim3 grid(B / BT), block(32);
    kalmannet_gru_scan<<<grid, block, 0, stream>>>(
        inputs, ln_w, ln_b, w_ih, w_hh, b_ih, b_hh, fc_w, fc_b, out);
}
